// _FAHead_2894807957850
// MI455X (gfx1250) — matrix operand order repaired and checked by reading, not compiled, not run on hardware
//
#include <hip/hip_runtime.h>
#include <hip/hip_bf16.h>
#include <stdint.h>

typedef __attribute__((ext_vector_type(16))) _Float16 v16h;
typedef __attribute__((ext_vector_type(8)))  float    v8f;

#define BN_EPS 1e-5f

// ---------------------------------------------------------------------------
// K0a: elementwise f32 -> f16 conversion (grid-stride)
// ---------------------------------------------------------------------------
__global__ void k_f32_to_f16(const float* __restrict__ src,
                             _Float16* __restrict__ dst, int n) {
  int i = blockIdx.x * blockDim.x + threadIdx.x;
  int stride = gridDim.x * blockDim.x;
  for (; i < n; i += stride) dst[i] = (_Float16)src[i];
}

// ---------------------------------------------------------------------------
// K0b: fold eval-mode BatchNorm into per-channel scale/shift
// ---------------------------------------------------------------------------
__global__ void k_bn_coeffs(const float* __restrict__ g, const float* __restrict__ b,
                            const float* __restrict__ m, const float* __restrict__ v,
                            float* __restrict__ scale, float* __restrict__ shift, int C) {
  int i = blockIdx.x * blockDim.x + threadIdx.x;
  if (i < C) {
    float s = g[i] * rsqrtf(v[i] + BN_EPS);
    scale[i] = s;
    shift[i] = b[i] - m[i] * s;
  }
}

// ---------------------------------------------------------------------------
// WMMA GEMM:  D[b] = A[b] (MxK) * B[b] (KxN  or, if B_TRANSPOSED, NxK -> A*B^T)
// Workgroup: 256 threads (8 waves as 2(M) x 4(N)), block tile 64(M) x 256(N),
// K-step 32.  Wave tile: 32(M) x 64(N) -> 2 A-frags x 4 B-frags -> 8 v_wmma
// per K-step (8 WMMAs per 12 ds_load_b128).
// Software pipeline: double-buffered LDS; next K-tile global loads issued
// before the current tile's WMMAs, ds_store after, one barrier per step.
// Epilogue MODE:
//   0: y=relu(acc*scale[c]+shift[c]) -> out32 and out16         (feat_x CBR)
//   1: y=relu(acc*scale[c]+shift[c]) -> out16                   (feat_f CBR)
//   2: raw acc -> out32                                         (att logits)
//   3: y=alpha*acc + aux32[b,c,p]    -> out16                   (feat_e + residual)
//   4: y=relu(acc*scale[c]+shift[c]) -> out32                   (final CBR)
// ---------------------------------------------------------------------------
#define LDSPITCH 40  // f16 elements per LDS row (20 dwords -> conflict-free frag reads)

template <int MODE, bool B_TRANSPOSED>
__global__ __launch_bounds__(256) void k_gemm_wmma(
    const _Float16* __restrict__ A, const _Float16* __restrict__ B,
    int M, int N, int K,
    long long strideA, long long strideB, long long strideD,
    float* __restrict__ out32, _Float16* __restrict__ out16,
    const float* __restrict__ scale, const float* __restrict__ shift,
    const float* __restrict__ aux32, long long strideAux,
    const float* __restrict__ alphaPtr) {
  __shared__ __align__(16) _Float16 sA[2][64][LDSPITCH];    // 2 x  5 KB
  __shared__ __align__(16) _Float16 sB[2][256][LDSPITCH];   // 2 x 20 KB

  const int tid   = threadIdx.x;
  const int lane  = tid & 31;
  const int wave  = tid >> 5;
  const int waveM = wave & 1;   // 2 waves along M (2*32 = 64)
  const int waveN = wave >> 1;  // 4 waves along N (4*64 = 256)
  const int m0 = blockIdx.y * 64;
  const int n0 = blockIdx.x * 256;
  const int bz = blockIdx.z;

  const _Float16* Ab = A + (long long)bz * strideA;
  const _Float16* Bb = B + (long long)bz * strideB;

  v8f acc[2][4] = {};

  const int mLane = lane & 15;  // A row within 16 / B column within 16
  const int hLane = lane >> 4;  // half-wave selects K bank
  const int aoff  = 8 * hLane;  // A fragment dword bank offset
  const int boff  = 8 * hLane;  // B fragment dword bank offset

  // ---- staging assignments (fixed per thread) ----
  const int aRow = tid >> 2;          // 0..63
  const int aCol = (tid & 3) * 8;     // 0,8,16,24 (f16 elems)
  // B tile is 16 KB/step = 4 uint4 per thread.
  //  B_TRANSPOSED ([n][k]): 4 chunks, direct b128 copy.
  //  B_N ([k][n]): 2 pair-chunks; each loads k-rows {k2,k2+1} x 8 n and
  //  commits packed {k2,k2+1} dwords via v_perm_b32 -> 8 ds_store_b32.
  int bR0[4], bC0[4];
#pragma unroll
  for (int it = 0; it < 4; ++it) {
    int chunk = tid * 4 + it;                 // 0..1023
    bR0[it] = chunk >> 2;                     // n row 0..255
    bC0[it] = (chunk & 3) * 8;                // k col 0,8,16,24
  }
  int pK2[2], pN0[2];
#pragma unroll
  for (int it = 0; it < 2; ++it) {
    int c2 = tid * 2 + it;                    // 0..511
    pK2[it] = (c2 >> 5) * 2;                  // even k row 0..30
    pN0[it] = (c2 & 31) * 8;                  // n offset 0..248
  }

  auto loadTile = [&](int k0, uint4& aR, uint4* bR) {
    aR = *(const uint4*)(Ab + (long long)(m0 + aRow) * K + k0 + aCol);
    if (B_TRANSPOSED) {
#pragma unroll
      for (int it = 0; it < 4; ++it)
        bR[it] = *(const uint4*)(Bb + (long long)(n0 + bR0[it]) * K + k0 + bC0[it]);
    } else {
#pragma unroll
      for (int it = 0; it < 2; ++it) {
        bR[2 * it + 0] = *(const uint4*)(Bb + (long long)(k0 + pK2[it] + 0) * N + n0 + pN0[it]);
        bR[2 * it + 1] = *(const uint4*)(Bb + (long long)(k0 + pK2[it] + 1) * N + n0 + pN0[it]);
      }
    }
  };
  auto storeTile = [&](int q, const uint4& aR, const uint4* bR) {
    *(uint4*)&sA[q][aRow][aCol] = aR;
    if (B_TRANSPOSED) {
#pragma unroll
      for (int it = 0; it < 4; ++it)
        *(uint4*)&sB[q][bR0[it]][bC0[it]] = bR[it];
    } else {
#pragma unroll
      for (int it = 0; it < 2; ++it) {
        const uint32_t* lo = (const uint32_t*)&bR[2 * it + 0];  // k = pK2   (4 dwords)
        const uint32_t* hi = (const uint32_t*)&bR[2 * it + 1];  // k = pK2+1 (4 dwords)
#pragma unroll
        for (int j = 0; j < 8; ++j) {
          // one v_perm_b32 packs {lo16,hi16} -> dword holding k=pK2, pK2+1
          uint32_t w = __builtin_amdgcn_perm(
              hi[j >> 1], lo[j >> 1],
              (j & 1) ? 0x07060302u : 0x05040100u);
          *(uint32_t*)&sB[q][pN0[it] + j][pK2[it]] = w;
        }
      }
    }
  };

  // ---- prologue: stage tile 0 into buffer 0 ----
  uint4 aR;
  uint4 bR[4];
  loadTile(0, aR, bR);
  storeTile(0, aR, bR);
  __syncthreads();

  int p = 0;
  for (int k0 = 0; k0 < K; k0 += 32) {
    const bool hasNext = (k0 + 32) < K;
    // (1) issue next tile's global loads (fly during the WMMAs below)
    if (hasNext) loadTile(k0 + 32, aR, bR);
    if (k0 + 64 < K) {  // speculative prefetch 2 tiles ahead
      __builtin_prefetch(Ab + (long long)(m0 + aRow) * K + k0 + 64, 0, 1);
      if (B_TRANSPOSED)
        __builtin_prefetch(Bb + (long long)(n0 + bR0[0]) * K + k0 + 64, 0, 1);
      else
        __builtin_prefetch(Bb + (long long)(k0 + 64 + pK2[0]) * N + n0, 0, 1);
    }

    // (2) fragments + 8 WMMAs (each B-frag feeds 2 M-subtiles)
    union { v16h v; uint32_t u[8]; } af[2], bf;
#pragma unroll
    for (int i = 0; i < 2; ++i) {
      const uint32_t* sa = (const uint32_t*)&sA[p][waveM * 32 + i * 16 + mLane][0];
#pragma unroll
      for (int r = 0; r < 8; ++r) af[i].u[r] = sa[r + aoff];
    }
#pragma unroll
    for (int j = 0; j < 4; ++j) {
      const uint32_t* sb = (const uint32_t*)&sB[p][waveN * 64 + j * 16 + mLane][0];
#pragma unroll
      for (int r = 0; r < 8; ++r) bf.u[r] = sb[boff + r];
#pragma unroll
      for (int i = 0; i < 2; ++i)
        acc[i][j] = __builtin_amdgcn_wmma_f32_16x16x32_f16(
            false, af[i].v, false, bf.v, (short)0, acc[i][j], false, false);
    }

    // (3) commit next tile into the other buffer, then one barrier
    if (hasNext) storeTile(p ^ 1, aR, bR);
    __syncthreads();
    p ^= 1;
  }

  // ---- epilogue; C/D layout: lane n = lane&15, VGPR r -> M = r + 8*(lane>>4)
  float alpha = 0.0f;
  if (MODE == 3) alpha = alphaPtr[0];
#pragma unroll
  for (int i = 0; i < 2; ++i) {
#pragma unroll
    for (int j = 0; j < 4; ++j) {
      int col = n0 + waveN * 64 + j * 16 + mLane;
#pragma unroll
      for (int r = 0; r < 8; ++r) {
        int row = m0 + waveM * 32 + i * 16 + r + 8 * hLane;
        float v = acc[i][j][r];
        long long idx = (long long)bz * strideD + (long long)row * N + col;
        if (MODE == 0) {
          float y = fmaxf(v * scale[row] + shift[row], 0.0f);
          out32[idx] = y;
          out16[idx] = (_Float16)y;
        } else if (MODE == 1) {
          out16[idx] = (_Float16)fmaxf(v * scale[row] + shift[row], 0.0f);
        } else if (MODE == 2) {
          out32[idx] = v;
        } else if (MODE == 3) {
          float y = alpha * v + aux32[(long long)bz * strideAux + (long long)row * N + col];
          out16[idx] = (_Float16)y;
        } else {
          out32[idx] = fmaxf(v * scale[row] + shift[row], 0.0f);
        }
      }
    }
  }
}

// ---------------------------------------------------------------------------
// Softmax of (rowmax - a) over last dim == exp(rowmin - a)/sum (shift cancels)
// One 128-thread block per row of length C=512; writes f16 for the next WMMA.
// ---------------------------------------------------------------------------
__global__ __launch_bounds__(128) void k_softmax_inv(
    const float* __restrict__ att, _Float16* __restrict__ out, int C) {
  long long rowBase = (long long)blockIdx.x * C;
  __shared__ float red[128];
  int tid = threadIdx.x;

  float vals[4];
  float mn = 3.4e38f;
#pragma unroll
  for (int i = 0; i < 4; ++i) {
    vals[i] = att[rowBase + tid + i * 128];
    mn = fminf(mn, vals[i]);
  }
  red[tid] = mn;
  __syncthreads();
  for (int s = 64; s > 0; s >>= 1) {
    if (tid < s) red[tid] = fminf(red[tid], red[tid + s]);
    __syncthreads();
  }
  float mnAll = red[0];
  __syncthreads();

  float e[4], sum = 0.0f;
#pragma unroll
  for (int i = 0; i < 4; ++i) {
    e[i] = __expf(mnAll - vals[i]);  // arg <= 0: stable
    sum += e[i];
  }
  red[tid] = sum;
  __syncthreads();
  for (int s = 64; s > 0; s >>= 1) {
    if (tid < s) red[tid] += red[tid + s];
    __syncthreads();
  }
  float inv = 1.0f / red[0];
#pragma unroll
  for (int i = 0; i < 4; ++i)
    out[rowBase + tid + i * 128] = (_Float16)(e[i] * inv);
}

// ---------------------------------------------------------------------------
extern "C" void kernel_launch(void* const* d_in, const int* in_sizes, int n_in,
                              void* d_out, int out_size, void* d_ws, size_t ws_size,
                              hipStream_t stream) {
  (void)in_sizes; (void)n_in; (void)out_size; (void)ws_size;
  constexpr int B = 8, Cin = 64, C = 512, HW = 64 * 64;  // n = 4096
  constexpr long long SXB = (long long)C * HW;    // per-batch feat stride (C x n)
  constexpr long long SFB = (long long)Cin * HW;  // per-batch fre stride
  constexpr long long SAT = (long long)C * C;     // per-batch att stride

  const float* x    = (const float*)d_in[0];
  const float* fre  = (const float*)d_in[1];
  const float* W_x1 = (const float*)d_in[2];
  const float* W_f1 = (const float*)d_in[3];
  const float* W_p2 = (const float*)d_in[4];
  const float* alphaPtr = (const float*)d_in[17];

  // workspace carve-up (256B aligned)
  char* ws = (char*)d_ws;
  size_t off = 0;
  auto take = [&](size_t bytes) -> char* {
    char* p = ws + off;
    off = (off + bytes + 255) & ~(size_t)255;
    return p;
  };
  _Float16* xh      = (_Float16*)take((size_t)B * SXB * 2);   // 33.5 MB
  _Float16* freh    = (_Float16*)take((size_t)B * SFB * 2);   //  4.2 MB
  _Float16* w1h     = (_Float16*)take((size_t)C * C * 2);
  _Float16* wfh     = (_Float16*)take((size_t)C * Cin * 2);
  _Float16* wph     = (_Float16*)take((size_t)C * C * 2);
  float*    featx32 = (float*)   take((size_t)B * SXB * 4);   // 67.1 MB
  _Float16* featxh  = (_Float16*)take((size_t)B * SXB * 2);   // 33.5 MB
  _Float16* featfh  = (_Float16*)take((size_t)B * SXB * 2);   // 33.5 MB
  float*    att32   = (float*)   take((size_t)B * SAT * 4);   //  8.4 MB
  _Float16* atth    = (_Float16*)take((size_t)B * SAT * 2);   //  4.2 MB
  _Float16* tmph    = (_Float16*)take((size_t)B * SXB * 2);   // 33.5 MB
  float* sc_x1 = (float*)take(C * 4); float* sh_x1 = (float*)take(C * 4);
  float* sc_f1 = (float*)take(C * 4); float* sh_f1 = (float*)take(C * 4);
  float* sc_p2 = (float*)take(C * 4); float* sh_p2 = (float*)take(C * 4);

  // K0: conversions + BN folding
  k_f32_to_f16<<<2048, 256, 0, stream>>>(x,    xh,   (int)(B * SXB));
  k_f32_to_f16<<<1024, 256, 0, stream>>>(fre,  freh, (int)(B * SFB));
  k_f32_to_f16<<<256,  256, 0, stream>>>(W_x1, w1h,  C * C);
  k_f32_to_f16<<<64,   256, 0, stream>>>(W_f1, wfh,  C * Cin);
  k_f32_to_f16<<<256,  256, 0, stream>>>(W_p2, wph,  C * C);
  k_bn_coeffs<<<2, 256, 0, stream>>>((const float*)d_in[5],  (const float*)d_in[6],
                                     (const float*)d_in[7],  (const float*)d_in[8],
                                     sc_x1, sh_x1, C);
  k_bn_coeffs<<<2, 256, 0, stream>>>((const float*)d_in[9],  (const float*)d_in[10],
                                     (const float*)d_in[11], (const float*)d_in[12],
                                     sc_f1, sh_f1, C);
  k_bn_coeffs<<<2, 256, 0, stream>>>((const float*)d_in[13], (const float*)d_in[14],
                                     (const float*)d_in[15], (const float*)d_in[16],
                                     sc_p2, sh_p2, C);

  // K1: feat_x = relu(bn(W_x1 @ x))   [512x512]x[512x4096] per batch
  {
    dim3 grid(HW / 256, C / 64, B);
    k_gemm_wmma<0, false><<<grid, 256, 0, stream>>>(
        w1h, xh, C, HW, C, 0, SXB, SXB,
        featx32, featxh, sc_x1, sh_x1, nullptr, 0, nullptr);
  }
  // K2: feat_f = relu(bn(W_f1 @ fre)) [512x64]x[64x4096] per batch
  {
    dim3 grid(HW / 256, C / 64, B);
    k_gemm_wmma<1, false><<<grid, 256, 0, stream>>>(
        wfh, freh, C, HW, Cin, 0, SFB, SXB,
        nullptr, featfh, sc_f1, sh_f1, nullptr, 0, nullptr);
  }
  // K3: att = fa @ ff^T   M=N=512, K=4096 (B given [N][K] -> NT path)
  {
    dim3 grid(C / 256, C / 64, B);
    k_gemm_wmma<2, true><<<grid, 256, 0, stream>>>(
        featxh, featfh, C, C, HW, SXB, SXB, SAT,
        att32, nullptr, nullptr, nullptr, nullptr, 0, nullptr);
  }
  // K4: inverted-max softmax rows, emit f16
  k_softmax_inv<<<B * C, 128, 0, stream>>>(att32, atth, C);

  // K5: tmp = alpha*(att @ fa) + feat_x   [512x512]x[512x4096] per batch
  {
    dim3 grid(HW / 256, C / 64, B);
    k_gemm_wmma<3, false><<<grid, 256, 0, stream>>>(
        atth, featxh, C, HW, C, SAT, SXB, SXB,
        nullptr, tmph, nullptr, nullptr, featx32, SXB, alphaPtr);
  }
  // K6: out = relu(bn(W_p2 @ tmp))
  {
    dim3 grid(HW / 256, C / 64, B);
    k_gemm_wmma<4, false><<<grid, 256, 0, stream>>>(
        wph, tmph, C, HW, C, 0, SXB, SXB,
        (float*)d_out, nullptr, sc_p2, sh_p2, nullptr, 0, nullptr);
  }
}
